// manyToManyGRU_67894843015192
// MI455X (gfx1250) — matrix operand order repaired and checked by reading, not compiled, not run on hardware
//
#include <hip/hip_runtime.h>
#include <hip/hip_bf16.h>

#define DIN   1024
#define DOUT  1024
#define BATCH 32
#define SEQ   128
#define MBIG  (BATCH * SEQ)   // 4096 rows of flattened x
#define KC    64              // K-chunk per LDS stage

typedef __attribute__((ext_vector_type(16))) __bf16 v16bf;
typedef __attribute__((ext_vector_type(8)))  __bf16 v8bf;
typedef __attribute__((ext_vector_type(8)))  float  v8f;

__device__ __forceinline__ float sigmoidf_(float x) {
  return 1.0f / (1.0f + __expf(-x));
}

// ---- WMMA fragment loaders (contiguous k = 16*half + s for A and B, LDS ld = 64 elems) ----
__device__ __forceinline__ v16bf load_frag_a(const __bf16* tile, int ks, int lane) {
  const int lr = lane & 15, half = lane >> 4;
  const v8bf lo = *(const v8bf*)(tile + lr * 64 + ks + half * 16);
  const v8bf hi = *(const v8bf*)(tile + lr * 64 + ks + half * 16 + 8);
  v16bf f;
#pragma unroll
  for (int i = 0; i < 8; ++i) { f[i] = lo[i]; f[i + 8] = hi[i]; }
  return f;
}
__device__ __forceinline__ v16bf load_frag_b(const __bf16* tile, int ks, int lane) {
  const int lr = lane & 15, half = lane >> 4;
  return *(const v16bf*)(tile + lr * 64 + ks + half * 16);
}

// ---- double-buffered 32x64 GEMM tile: out[m,n] = sum_k A[m,k]*B[n,k] ----
// Block = 128 threads = 4 waves. Wave w: mtile = w&1 (16 rows), npair = w>>1 (32 cols).
// Each wave owns two 16x16 f32 accumulators. Prefetch staging lives in NAMED uint4
// scalars (no arrays -> no address-taken -> no scratch spills).
__device__ __forceinline__ void tile_gemm32x64(const __bf16* __restrict__ Abase, int lda,
                                               const __bf16* __restrict__ Bbase, int ldb,
                                               int K, __bf16* As, __bf16* Bs,
                                               v8f& acc0, v8f& acc1) {
  const int tid   = threadIdx.x;
  const int lane  = tid & 31;
  const int wave  = tid >> 5;
  const int mtile = wave & 1;
  const int npair = wave >> 1;

  // per-thread slices: A tile 32xKC (4 thr/row, 16 elems), B tile 64xKC (2 thr/row, 32)
  const int arow = tid >> 2, ac = (tid & 3) * 16;
  const int brow = tid >> 1, bc = (tid & 1) * 32;
  const __bf16* agp = Abase + (size_t)arow * lda + ac;
  const __bf16* bgp = Bbase + (size_t)brow * ldb + bc;
  __bf16* asl = As + arow * 64 + ac;
  __bf16* bsl = Bs + brow * 64 + bc;

  uint4 a0, a1, b0, b1, b2, b3;

  // preload chunk 0 straight to LDS buffer 0
  a0 = ((const uint4*)agp)[0];
  a1 = ((const uint4*)agp)[1];
  b0 = ((const uint4*)bgp)[0];
  b1 = ((const uint4*)bgp)[1];
  b2 = ((const uint4*)bgp)[2];
  b3 = ((const uint4*)bgp)[3];
  ((uint4*)asl)[0] = a0;
  ((uint4*)asl)[1] = a1;
  ((uint4*)bsl)[0] = b0;
  ((uint4*)bsl)[1] = b1;
  ((uint4*)bsl)[2] = b2;
  ((uint4*)bsl)[3] = b3;
  __syncthreads();

  int cur = 0;
  for (int kc = KC; kc <= K; kc += KC) {
    const bool more = (kc < K);
    if (more) {  // issue next chunk's global loads; they fly under the WMMAs below
      const uint4* ap = (const uint4*)(agp + kc);
      a0 = ap[0];
      a1 = ap[1];
      const uint4* bp = (const uint4*)(bgp + kc);
      b0 = bp[0];
      b1 = bp[1];
      b2 = bp[2];
      b3 = bp[3];
    }

    const __bf16* at  = As + cur * (32 * 64) + mtile * 16 * 64;
    const __bf16* bt0 = Bs + cur * (64 * 64) + (npair * 2) * 16 * 64;
    const __bf16* bt1 = bt0 + 16 * 64;
#pragma unroll
    for (int ks = 0; ks < KC; ks += 32) {
      v16bf a  = load_frag_a(at, ks, lane);
      v16bf f0 = load_frag_b(bt0, ks, lane);
      v16bf f1 = load_frag_b(bt1, ks, lane);
      acc0 = __builtin_amdgcn_wmma_f32_16x16x32_bf16(false, a, false, f0,
                                                     (short)0, acc0, false, false);
      acc1 = __builtin_amdgcn_wmma_f32_16x16x32_bf16(false, a, false, f1,
                                                     (short)0, acc1, false, false);
    }

    if (more) {  // spill staged regs into the other LDS buffer
      __bf16* as2 = asl + (cur ^ 1) * (32 * 64);
      __bf16* bs2 = bsl + (cur ^ 1) * (64 * 64);
      ((uint4*)as2)[0] = a0;
      ((uint4*)as2)[1] = a1;
      ((uint4*)bs2)[0] = b0;
      ((uint4*)bs2)[1] = b1;
      ((uint4*)bs2)[2] = b2;
      ((uint4*)bs2)[3] = b3;
    }
    __syncthreads();
    cur ^= 1;
  }
}

// ---------------- utility kernels ----------------
__global__ void cvt_f32_bf16_kernel(const float* __restrict__ src,
                                    __bf16* __restrict__ dst, int n) {
  int i = blockIdx.x * blockDim.x + threadIdx.x;
  if (i < n) dst[i] = (__bf16)src[i];
}

__global__ void init_h_kernel(float* __restrict__ hf, __bf16* __restrict__ hb, int n) {
  int i = blockIdx.x * blockDim.x + threadIdx.x;
  if (i < n) { hf[i] = 0.0f; hb[i] = (__bf16)0.0f; }
}

// ---------------- phase 1: Ax[g] = x @ Wg^T + (bg + cg), g in {z,r,h} ----------------
__global__ void __launch_bounds__(128)
gemm_x_kernel(const __bf16* __restrict__ xbf,
              const __bf16* __restrict__ Wz, const __bf16* __restrict__ Wr,
              const __bf16* __restrict__ Wh,
              const float* __restrict__ bz, const float* __restrict__ cz,
              const float* __restrict__ br, const float* __restrict__ cr,
              const float* __restrict__ bh, const float* __restrict__ ch,
              float* __restrict__ Ax) {
  __shared__ __align__(64) __bf16 As[2 * 32 * 64];
  __shared__ __align__(64) __bf16 Bs[2 * 64 * 64];
  const int nbase = blockIdx.x * 64;
  const int mbase = blockIdx.y * 32;
  const int gate  = blockIdx.z;
  const __bf16* W = (gate == 0) ? Wz : (gate == 1) ? Wr : Wh;
  const float*  b = (gate == 0) ? bz : (gate == 1) ? br : bh;
  const float*  c = (gate == 0) ? cz : (gate == 1) ? cr : ch;

  v8f acc0 = {0.f,0.f,0.f,0.f,0.f,0.f,0.f,0.f};
  v8f acc1 = acc0;
  tile_gemm32x64(xbf + (size_t)mbase * DIN, DIN,
                 W + (size_t)nbase * DIN, DIN, DIN, As, Bs, acc0, acc1);

  const int lane = threadIdx.x & 31, wave = threadIdx.x >> 5;
  const int mtile = wave & 1, npair = wave >> 1;
  const int half = lane >> 4, nl = lane & 15;
  const int n0 = nbase + npair * 32 + nl;
  const int n1 = n0 + 16;
  const float bias0 = b[n0] + c[n0];
  const float bias1 = b[n1] + c[n1];
  float* out = Ax + (size_t)gate * ((size_t)MBIG * DOUT);
#pragma unroll
  for (int v = 0; v < 8; ++v) {
    const int m = mbase + mtile * 16 + half * 8 + v;
    out[(size_t)m * DOUT + n0] = acc0[v] + bias0;
    out[(size_t)m * DOUT + n1] = acc1[v] + bias1;
  }
}

// ---------------- phase 2a: z_t and r_t (gridDim.y selects gate) ----------------
__global__ void __launch_bounds__(128)
step_zr_kernel(const __bf16* __restrict__ hbf,
               const __bf16* __restrict__ Uz, const __bf16* __restrict__ Ur,
               const float* __restrict__ Ax,     // 3-gate base
               const float* __restrict__ hf32,
               float* __restrict__ zbuf, __bf16* __restrict__ rhbf, int t) {
  __shared__ __align__(64) __bf16 As[2 * 32 * 64];
  __shared__ __align__(64) __bf16 Bs[2 * 64 * 64];
  const int nbase = blockIdx.x * 64;
  const int gate  = blockIdx.y;   // 0 = z, 1 = r
  const __bf16* U = (gate == 0) ? Uz : Ur;

  v8f acc0 = {0.f,0.f,0.f,0.f,0.f,0.f,0.f,0.f};
  v8f acc1 = acc0;
  tile_gemm32x64(hbf, DOUT, U + (size_t)nbase * DOUT, DOUT, DOUT, As, Bs, acc0, acc1);

  const int lane = threadIdx.x & 31, wave = threadIdx.x >> 5;
  const int mtile = wave & 1, npair = wave >> 1;
  const int half = lane >> 4, nl = lane & 15;
  const int n0 = nbase + npair * 32 + nl;
  const int n1 = n0 + 16;
  const float* axg = Ax + (size_t)gate * ((size_t)MBIG * DOUT);
#pragma unroll
  for (int v = 0; v < 8; ++v) {
    const int m = mtile * 16 + half * 8 + v;            // batch index
    const size_t axrow = ((size_t)(m * SEQ + t)) * DOUT;
    const float s0 = sigmoidf_(acc0[v] + axg[axrow + n0]);
    const float s1 = sigmoidf_(acc1[v] + axg[axrow + n1]);
    if (gate == 0) {
      zbuf[m * DOUT + n0] = s0;
      zbuf[m * DOUT + n1] = s1;
    } else {
      rhbf[m * DOUT + n0] = (__bf16)(s0 * hf32[m * DOUT + n0]);
      rhbf[m * DOUT + n1] = (__bf16)(s1 * hf32[m * DOUT + n1]);
    }
  }
}

// ---------------- phase 2b: h_hat, h update, emit output row ----------------
__global__ void __launch_bounds__(128)
step_h_kernel(const __bf16* __restrict__ rhbf, const __bf16* __restrict__ Uh,
              const float* __restrict__ Axh,   // gate-2 base
              const float* __restrict__ zbuf,
              float* __restrict__ hf32, __bf16* __restrict__ hbf,
              float* __restrict__ out, int t) {
  __shared__ __align__(64) __bf16 As[2 * 32 * 64];
  __shared__ __align__(64) __bf16 Bs[2 * 64 * 64];
  const int nbase = blockIdx.x * 64;

  v8f acc0 = {0.f,0.f,0.f,0.f,0.f,0.f,0.f,0.f};
  v8f acc1 = acc0;
  tile_gemm32x64(rhbf, DOUT, Uh + (size_t)nbase * DOUT, DOUT, DOUT, As, Bs, acc0, acc1);

  const int lane = threadIdx.x & 31, wave = threadIdx.x >> 5;
  const int mtile = wave & 1, npair = wave >> 1;
  const int half = lane >> 4, nl = lane & 15;
  const int n0 = nbase + npair * 32 + nl;
  const int n1 = n0 + 16;
#pragma unroll
  for (int v = 0; v < 8; ++v) {
    const int m = mtile * 16 + half * 8 + v;            // batch index
    const size_t axrow = ((size_t)(m * SEQ + t)) * DOUT;
#pragma unroll
    for (int q = 0; q < 2; ++q) {
      const int n = (q == 0) ? n0 : n1;
      const float a = (q == 0) ? acc0[v] : acc1[v];
      const float hh   = sigmoidf_(a + Axh[axrow + n]);
      const float z    = zbuf[m * DOUT + n];
      const float hold = hf32[m * DOUT + n];
      const float hn   = (1.0f - z) * hold + z * hh;
      hf32[m * DOUT + n] = hn;
      hbf[m * DOUT + n]  = (__bf16)hn;
      // reference stores (T,B,D) then reinterprets as (B,T,D): flat = (t*B+b)*D+n
      out[((size_t)(t * BATCH + m)) * DOUT + n] = hn;
    }
  }
}

extern "C" void kernel_launch(void* const* d_in, const int* in_sizes, int n_in,
                              void* d_out, int out_size, void* d_ws, size_t ws_size,
                              hipStream_t stream) {
  const float* x  = (const float*)d_in[0];
  const float* Wz = (const float*)d_in[1];
  const float* bz = (const float*)d_in[2];
  const float* Uz = (const float*)d_in[3];
  const float* cz = (const float*)d_in[4];
  const float* Wr = (const float*)d_in[5];
  const float* br = (const float*)d_in[6];
  const float* Ur = (const float*)d_in[7];
  const float* cr = (const float*)d_in[8];
  const float* Wh = (const float*)d_in[9];
  const float* bh = (const float*)d_in[10];
  const float* Uh = (const float*)d_in[11];
  const float* ch = (const float*)d_in[12];
  float* out = (float*)d_out;

  // workspace carve-up (256B aligned regions)
  char* ws = (char*)d_ws;
  size_t off = 0;
  auto carve = [&](size_t bytes) -> void* {
    void* p = ws + off;
    off = (off + bytes + 255) & ~(size_t)255;
    return p;
  };
  __bf16* xbf  = (__bf16*)carve((size_t)MBIG * DIN * 2);
  __bf16* Wzb  = (__bf16*)carve((size_t)DOUT * DIN * 2);
  __bf16* Wrb  = (__bf16*)carve((size_t)DOUT * DIN * 2);
  __bf16* Whb  = (__bf16*)carve((size_t)DOUT * DIN * 2);
  __bf16* Uzb  = (__bf16*)carve((size_t)DOUT * DOUT * 2);
  __bf16* Urb  = (__bf16*)carve((size_t)DOUT * DOUT * 2);
  __bf16* Uhb  = (__bf16*)carve((size_t)DOUT * DOUT * 2);
  float*  Ax   = (float*)carve((size_t)3 * MBIG * DOUT * 4);
  float*  hf32 = (float*)carve((size_t)BATCH * DOUT * 4);
  __bf16* hbf  = (__bf16*)carve((size_t)BATCH * DOUT * 2);
  float*  zbuf = (float*)carve((size_t)BATCH * DOUT * 4);
  __bf16* rhbf = (__bf16*)carve((size_t)BATCH * DOUT * 2);

  // precision conversions (weights: 12MB bf16 total -> resident in 192MB L2)
  {
    int n = MBIG * DIN;
    cvt_f32_bf16_kernel<<<(n + 255) / 256, 256, 0, stream>>>(x, xbf, n);
    n = DOUT * DIN;
    cvt_f32_bf16_kernel<<<(n + 255) / 256, 256, 0, stream>>>(Wz, Wzb, n);
    cvt_f32_bf16_kernel<<<(n + 255) / 256, 256, 0, stream>>>(Wr, Wrb, n);
    cvt_f32_bf16_kernel<<<(n + 255) / 256, 256, 0, stream>>>(Wh, Whb, n);
    cvt_f32_bf16_kernel<<<(n + 255) / 256, 256, 0, stream>>>(Uz, Uzb, n);
    cvt_f32_bf16_kernel<<<(n + 255) / 256, 256, 0, stream>>>(Ur, Urb, n);
    cvt_f32_bf16_kernel<<<(n + 255) / 256, 256, 0, stream>>>(Uh, Uhb, n);
  }
  {
    int n = BATCH * DOUT;
    init_h_kernel<<<(n + 255) / 256, 256, 0, stream>>>(hf32, hbf, n);
  }

  // phase 1: all x-projections for all gates in one big WMMA GEMM
  gemm_x_kernel<<<dim3(DOUT / 64, MBIG / 32, 3), 128, 0, stream>>>(
      xbf, Wzb, Wrb, Whb, bz, cz, br, cr, bh, ch, Ax);

  // phase 2: sequential scan, 2 launches per step (global sync needed before Uh GEMM)
  for (int t = 0; t < SEQ; ++t) {
    step_zr_kernel<<<dim3(DOUT / 64, 2), 128, 0, stream>>>(
        hbf, Uzb, Urb, Ax, hf32, zbuf, rhbf, t);
    step_h_kernel<<<dim3(DOUT / 64, 1), 128, 0, stream>>>(
        rhbf, Uhb, Ax + (size_t)2 * MBIG * DOUT, zbuf, hf32, hbf, out, t);
  }
}
